// TemporalReadout_37237366456838
// MI455X (gfx1250) — matrix operand order repaired and checked by reading, not compiled, not run on hardware
//
#include <hip/hip_runtime.h>

// GRU (B=32, T=256, C=64, H=128) + FC(P=10), fused single kernel.
// Grid: 32 blocks (one per batch b). Block: 256 threads = 8 waves (wave32).
// Each wave owns a 16-wide slice of H and computes all three gates for it.
// r/z gates accumulate both GEMMs (x*Wih and h*Whh) into a single WMMA
// accumulator with biases preloaded; only the n gate keeps xn/hn separate.
// Double buffers are selected by integer offset from a fixed LDS base so the
// compiler keeps AS3 provenance (ds_load, not flat_load).

#define T_ 256
#define C_ 64
#define H_ 128
#define P_ 10
#define G3_ 384
#define WSTRIDE 136   // bf16 row stride (weights, x tiles, h buffers): 272B, bank-skewed
#define HLSTRIDE 132  // f32 row stride for final-h scratch
#define BUFELTS (C_ * WSTRIDE)   // one x/h buffer, in bf16 elements

#define AS1 __attribute__((address_space(1)))

typedef __attribute__((ext_vector_type(16))) __bf16 bf16x16;
typedef __attribute__((ext_vector_type(8)))  __bf16 bf16x8;
typedef __attribute__((ext_vector_type(4)))  __bf16 bf16x4;
typedef __attribute__((ext_vector_type(8)))  float  f32x8;
typedef __attribute__((ext_vector_type(4)))  float  f32x4;

typedef const AS1 float* gcf;
typedef AS1 float*       gf;

static __device__ __forceinline__ gcf as_gc(const float* p) {
  return (gcf)(unsigned long long)p;
}
static __device__ __forceinline__ gf as_g(float* p) {
  return (gf)(unsigned long long)p;
}

static __device__ __forceinline__ f32x8 wmma_bf16(bf16x16 a, bf16x16 b, f32x8 c) {
  // 8 args: (neg_a, A, neg_b, B, c_mod, C, reuse_a, reuse_b)
  return __builtin_amdgcn_wmma_f32_16x16x32_bf16(false, a, false, b, (short)0, c,
                                                 false, false);
}

static __device__ __forceinline__ bf16x16 cat8(bf16x8 lo, bf16x8 hi) {
  bf16x16 r;
#pragma unroll
  for (int i = 0; i < 8; ++i) { r[i] = lo[i]; r[i + 8] = hi[i]; }
  return r;
}

// A-matrix fragment (16x32 bf16) from a bf16 LDS tile, row-major stride WSTRIDE.
// Lanes 0-15: M=lane, K = {kk*32+0..7, kk*32+16..23}; lanes 16-31: M=lane-16, K+8.
static __device__ __forceinline__ bf16x16 ldsA(const __bf16* base, int off,
                                               int row0, int kk, int lane) {
  int m = row0 + (lane & 15);
  int kbase = kk * 32 + ((lane & 16) ? 8 : 0);
  const __bf16* p = base + off + m * WSTRIDE + kbase;
  return cat8(*(const bf16x8*)p, *(const bf16x8*)(p + 16));
}

// B-matrix fragment (32x16 bf16): lane holds column n = ncol0 + (lane&15),
// K = kk*32 + {0..7, 16..23} (lanes 16-31: +8). W stored row-major (gate-row x K).
static __device__ __forceinline__ bf16x16 ldsB(const __bf16* w, int ncol0,
                                               int kk, int lane) {
  int n = ncol0 + (lane & 15);
  int kbase = kk * 32 + ((lane & 16) ? 8 : 0);
  const __bf16* p = w + n * WSTRIDE + kbase;
  return cat8(*(const bf16x8*)p, *(const bf16x8*)(p + 16));
}

static __device__ __forceinline__ float sigmoidf_(float x) {
  return __builtin_amdgcn_rcpf(1.0f + __expf(-x));
}

static __device__ __forceinline__ float tanhf_(float x) {
#if __has_builtin(__builtin_amdgcn_tanhf)
  return __builtin_amdgcn_tanhf(x);
#else
  float xc = fminf(fmaxf(x, -15.0f), 15.0f);
  float e = __expf(-2.0f * xc);
  return (1.0f - e) * __builtin_amdgcn_rcpf(1.0f + e);
#endif
}

static __device__ __forceinline__ f32x8 splat8(float s) {
  f32x8 v = {s, s, s, s, s, s, s, s};
  return v;
}

__global__ __launch_bounds__(256)
void TemporalReadout_37237366456838_kernel(const float* __restrict__ Z,
                                           const float* __restrict__ w_ih,
                                           const float* __restrict__ w_hh,
                                           const float* __restrict__ b_ih,
                                           const float* __restrict__ b_hh,
                                           const float* __restrict__ fc_w,
                                           const float* __restrict__ fc_b,
                                           float* __restrict__ out) {
  extern __shared__ char smem[];
  __bf16* wih = (__bf16*)smem;                 // 384 x 136 bf16
  __bf16* whh = wih + G3_ * WSTRIDE;           // 384 x 136 bf16
  __bf16* xbb = whh + G3_ * WSTRIDE;           // 2 x (64 x 136) bf16, double buffer
  __bf16* hbb = xbb + 2 * BUFELTS;             // 2 x (64 x 136) bf16, double buffer

  const int tid  = threadIdx.x;
  const int lane = tid & 31;
  const int wave = tid >> 5;
  const int hs   = wave * 16;             // this wave's h-column slice
  const int hcol = hs + (lane & 15);
  const int b    = blockIdx.x;

  const float* Zb = Z + (size_t)b * T_ * C_ * H_;   // generic copy (for prefetch)
  gcf Zg    = as_gc(Zb);
  gcf wihg  = as_gc(w_ih);
  gcf whhg  = as_gc(w_hh);
  gcf bihg  = as_gc(b_ih);
  gcf bhhg  = as_gc(b_hh);
  gcf fcwg  = as_gc(fc_w);
  gcf fcbg  = as_gc(fc_b);
  gf  outg  = as_g(out);

  // ---- stage weights into LDS as bf16 (one-time) ----
  for (int i = tid; i < G3_ * H_; i += 256) {
    int r = i >> 7, c = i & 127;
    wih[r * WSTRIDE + c] = (__bf16)wihg[i];
    whh[r * WSTRIDE + c] = (__bf16)whhg[i];
  }
  // ---- h0 = 0 (buffer 0) ----
  for (int i = tid; i < BUFELTS; i += 256) hbb[i] = (__bf16)0.0f;
  // ---- stage X tile for t=0 (fp32 -> bf16, buffer 0) ----
#pragma unroll
  for (int k = 0; k < 8; ++k) {
    int ch = tid + (k << 8);
    int row = ch >> 5, col = (ch & 31) << 2;
    f32x4 v = *(const AS1 f32x4*)(Zg + row * H_ + col);
    bf16x4 w;
#pragma unroll
    for (int i = 0; i < 4; ++i) w[i] = (__bf16)v[i];
    *(bf16x4*)(xbb + row * WSTRIDE + col) = w;
  }

  // ---- per-lane gate biases, folded into WMMA accumulator init ----
  const float sR  = bihg[hcol]          + bhhg[hcol];
  const float sZ  = bihg[H_ + hcol]     + bhhg[H_ + hcol];
  const float sXN = bihg[2 * H_ + hcol];
  const float sHN = bhhg[2 * H_ + hcol];

  float h[4][8];
#pragma unroll
  for (int rt = 0; rt < 4; ++rt)
#pragma unroll
    for (int r = 0; r < 8; ++r) h[rt][r] = 0.0f;

  for (int t = 0; t < T_; ++t) {
    __syncthreads();   // hb/x writes from previous step visible

    // issue global loads for t+1 (converted + stored to LDS at end of step)
    f32x4 stg[8];
    const bool doStage = (t + 1 < T_);
    if (doStage) {
      gcf srcN = Zg + (size_t)(t + 1) * C_ * H_;
#pragma unroll
      for (int k = 0; k < 8; ++k) {
        int ch = tid + (k << 8);
        stg[k] = *(const AS1 f32x4*)(srcN + (ch >> 5) * H_ + ((ch & 31) << 2));
      }
      if (t + 2 < T_)  // one 128B line per thread covers the t+2 tile
        __builtin_prefetch(Zb + (size_t)(t + 2) * C_ * H_ + tid * 32, 0, 0);
    }

    // double-buffer selection by integer offset from fixed LDS bases
    const int cur = (t & 1) * BUFELTS;
    const int nxt = BUFELTS - cur;
    __bf16* hn = hbb + nxt;

#pragma unroll
    for (int rt = 0; rt < 4; ++rt) {
      bf16x16 ax[4], ah[4];
#pragma unroll
      for (int kk = 0; kk < 4; ++kk) {
        ax[kk] = ldsA(xbb, cur, rt * 16, kk, lane);
        ah[kk] = ldsA(hbb, cur, rt * 16, kk, lane);
      }
      f32x8 accR  = splat8(sR);
      f32x8 accZ  = splat8(sZ);
      f32x8 accXN = splat8(sXN);
      f32x8 accHN = splat8(sHN);
#pragma unroll
      for (int kk = 0; kk < 4; ++kk) {
        accR  = wmma_bf16(ax[kk], ldsB(wih, 0 * H_ + hs, kk, lane), accR);
        accR  = wmma_bf16(ah[kk], ldsB(whh, 0 * H_ + hs, kk, lane), accR);
        accZ  = wmma_bf16(ax[kk], ldsB(wih, 1 * H_ + hs, kk, lane), accZ);
        accZ  = wmma_bf16(ah[kk], ldsB(whh, 1 * H_ + hs, kk, lane), accZ);
        accXN = wmma_bf16(ax[kk], ldsB(wih, 2 * H_ + hs, kk, lane), accXN);
        accHN = wmma_bf16(ah[kk], ldsB(whh, 2 * H_ + hs, kk, lane), accHN);
      }
      // gate nonlinearity; D layout: VGPR r <-> row rt*16 + (lane<16?0:8)+r, col hcol
      const int rbase = rt * 16 + ((lane & 16) ? 8 : 0);
#pragma unroll
      for (int r = 0; r < 8; ++r) {
        float rg = sigmoidf_(accR[r]);
        float zg = sigmoidf_(accZ[r]);
        float ng = tanhf_(accXN[r] + rg * accHN[r]);
        float hv = (1.0f - zg) * ng + zg * h[rt][r];
        h[rt][r] = hv;
        hn[(rbase + r) * WSTRIDE + hcol] = (__bf16)hv;
      }
    }

    if (doStage) {
      __bf16* xd = xbb + nxt;
#pragma unroll
      for (int k = 0; k < 8; ++k) {
        int ch = tid + (k << 8);
        bf16x4 w;
#pragma unroll
        for (int i = 0; i < 4; ++i) w[i] = (__bf16)stg[k][i];
        *(bf16x4*)(xd + (ch >> 5) * WSTRIDE + ((ch & 31) << 2)) = w;
      }
    }
  }

  // ---- final FC: pred[c][p] = h_last[c] . fc_w[p] + fc_b[p]; out (B,P,C) ----
  __syncthreads();
  float* hl = (float*)xbb;  // reuse x double buffer as fp32 h_last (64 x 132)
#pragma unroll
  for (int rt = 0; rt < 4; ++rt) {
    const int rbase = rt * 16 + ((lane & 16) ? 8 : 0);
#pragma unroll
    for (int r = 0; r < 8; ++r) hl[(rbase + r) * HLSTRIDE + hcol] = h[rt][r];
  }
  __syncthreads();
  if (tid < C_) {
    const int c = tid;
#pragma unroll
    for (int p = 0; p < P_; ++p) {
      float acc = fcbg[p];
#pragma unroll 4
      for (int k = 0; k < H_; ++k) acc += hl[c * HLSTRIDE + k] * fcwg[p * H_ + k];
      outg[(size_t)b * (P_ * C_) + p * C_ + c] = acc;
    }
  }
}

extern "C" void kernel_launch(void* const* d_in, const int* in_sizes, int n_in,
                              void* d_out, int out_size, void* d_ws, size_t ws_size,
                              hipStream_t stream) {
  (void)in_sizes; (void)n_in; (void)out_size; (void)d_ws; (void)ws_size;
  const float* Z    = (const float*)d_in[0];
  const float* w_ih = (const float*)d_in[1];
  const float* w_hh = (const float*)d_in[2];
  const float* b_ih = (const float*)d_in[3];
  const float* b_hh = (const float*)d_in[4];
  const float* fc_w = (const float*)d_in[5];
  const float* fc_b = (const float*)d_in[6];
  float* out = (float*)d_out;

  const size_t smem =
      (size_t)2 * G3_ * WSTRIDE * sizeof(__bf16) +   // wih + whh
      (size_t)4 * C_ * WSTRIDE * sizeof(__bf16);     // x + h double buffers

  TemporalReadout_37237366456838_kernel<<<dim3(32), dim3(256), smem, stream>>>(
      Z, w_ih, w_hh, b_ih, b_hh, fc_w, fc_b, out);
}
